// infoFSM_74723841016094
// MI455X (gfx1250) — compile-verified, matrix operand order checked by reading, not run on hardware
//
#include <hip/hip_runtime.h>
#include <math.h>

typedef __attribute__((ext_vector_type(16))) _Float16 v16h;
typedef __attribute__((ext_vector_type(8)))  _Float16 v8h;
typedef __attribute__((ext_vector_type(8)))  float    v8f;
typedef __attribute__((ext_vector_type(4)))  float    v4f;

#define B_    128
#define N_    512
#define D_    512
#define ROWS  (B_ * N_)        // 65536 token rows
#define MTILE 32               // rows per workgroup (2 x 16-row WMMA tiles)

// f16 weight pack offsets (in halves) inside d_ws
#define OFF_WL 0
#define OFF_W1 (512 * 512)                 // 262144
#define OFF_W2 (OFF_W1 + 256 * 512)        // 393216
#define OFF_W3 (OFF_W2 + 128 * 256)        // 425984
#define TOT_W  (OFF_W3 + 128)              // 426112 halves (~852 KB)

// ---------------------------------------------------------------------------
// Prologue: convert all MLP weights f32 -> f16 into workspace (L2-resident).
// ---------------------------------------------------------------------------
__global__ __launch_bounds__(256) void convert_weights_f16(
    const float* __restrict__ wL, const float* __restrict__ w1,
    const float* __restrict__ w2, const float* __restrict__ w3,
    _Float16* __restrict__ pack)
{
    int i = blockIdx.x * blockDim.x + threadIdx.x;
    if (i >= TOT_W) return;
    float v;
    if (i < OFF_W1)      v = wL[i];
    else if (i < OFF_W2) v = w1[i - OFF_W1];
    else if (i < OFF_W3) v = w2[i - OFF_W2];
    else                 v = w3[i - OFF_W3];
    pack[i] = (_Float16)v;
}

__device__ __forceinline__ float gelu_exact(float x) {
    // erf form (torch nn.GELU default)
    return 0.5f * x * (1.0f + erff(x * 0.70710678118654752f));
}

// Build a 16-half WMMA fragment from two contiguous 16B chunks (ISA 7.12.2
// 16-bit A/B layout: halves [k0+kb .. +7] and [k0+16+kb .. +7]).
#define FRAG16(base, kb)                                                      \
    __builtin_shufflevector(*(const v8h*)((base) + (kb)),                     \
                            *(const v8h*)((base) + 16 + (kb)),                \
                            0,1,2,3,4,5,6,7,8,9,10,11,12,13,14,15)

// ---------------------------------------------------------------------------
// Fused MLP mask kernel: one 32-row tile per workgroup, 8 waves (wave32).
// LDS regions are aliased by lifetime to stay at 64 KB:
//   [0,32K)   : Xs (f16 input tile)      -> later x2s (layer-2 activations)
//   [32K,64K) : x1s (layer-1 activations)-> later x3s + currs
// ---------------------------------------------------------------------------
__global__ __launch_bounds__(256) void fused_mask_mlp(
    const float*    __restrict__ X,       // [ROWS, 512] f32
    const float*    __restrict__ prev_m,  // [ROWS] f32
    const float*    __restrict__ w3f,     // [128] f32 (last layer, exact)
    const _Float16* __restrict__ wpack,   // f16 weight pack in d_ws
    float*          __restrict__ outp,    // [ROWS, 512] f32
    int*            __restrict__ maskp,   // [ROWS] i32
    float*          __restrict__ currp)   // [ROWS] f32
{
    __shared__ __align__(16) unsigned char smem[65536];
    _Float16* Xs    = (_Float16*)(smem);            // [32*512] f16, 32 KB
    _Float16* x1s   = (_Float16*)(smem + 32768);    // [32*512] f16, 32 KB
    _Float16* x2s   = (_Float16*)(smem);            // [32*256] f16 (alias Xs)
    _Float16* x3s   = (_Float16*)(smem + 32768);    // [32*128] f16 (alias x1s)
    float*    currs = (float*)   (smem + 32768 + 8192); // [32] f32

    const int tid   = threadIdx.x;
    const int lane  = tid & 31;
    const int wave  = tid >> 5;
    const int rbase = blockIdx.x * MTILE;

    // ---- warm L2 with the weight pack (global_prefetch_b8) ---------------
    if (blockIdx.x < 64) {
        for (int off = tid * 64; off < TOT_W; off += 256 * 64 * 64)
            __builtin_prefetch(wpack + off, 0, 1);
    }

    // ---- stage X tile to LDS as f16 (vectorized) -------------------------
    {
        const v4f* src = (const v4f*)(X + (size_t)rbase * 512);
        for (int i = tid; i < (MTILE * 512) / 4; i += 256) {
            v4f v = src[i];
            _Float16* d = &Xs[i * 4];
            d[0] = (_Float16)v.x; d[1] = (_Float16)v.y;
            d[2] = (_Float16)v.z; d[3] = (_Float16)v.w;
        }
    }
    __syncthreads();

    // 16-bit A/B fragment layout (ISA 7.12.2)
    const int row  = lane & 15;          // row within a 16-row tile
    const int nsel = lane & 15;          // output column within a 16-col tile
    const int kb   = (lane < 16) ? 0 : 8;
    const int rsh  = (lane >> 4) << 3;   // accumulator row offset (0 or 8)

    // ---- Layer 1: [32,512] x W_L^T -> gelu -> x1s [32,512] ---------------
    {
        v8f acc[4][2] = {};
        for (int k0 = 0; k0 < 512; k0 += 32) {
            v16h a0 = FRAG16(&Xs[(row)      * 512 + k0], kb);
            v16h a1 = FRAG16(&Xs[(row + 16) * 512 + k0], kb);
#pragma unroll
            for (int t = 0; t < 4; ++t) {
                int n = wave * 64 + t * 16 + nsel;
                v16h b = FRAG16(wpack + OFF_WL + (size_t)n * 512 + k0, kb);
                acc[t][0] = __builtin_amdgcn_wmma_f32_16x16x32_f16(
                    false, a0, false, b, (short)0, acc[t][0], false, false);
                acc[t][1] = __builtin_amdgcn_wmma_f32_16x16x32_f16(
                    false, a1, false, b, (short)0, acc[t][1], false, false);
            }
        }
#pragma unroll
        for (int t = 0; t < 4; ++t) {
            int col = wave * 64 + t * 16 + nsel;
#pragma unroll
            for (int rt = 0; rt < 2; ++rt)
#pragma unroll
                for (int i = 0; i < 8; ++i)
                    x1s[(rt * 16 + i + rsh) * 512 + col] =
                        (_Float16)gelu_exact(acc[t][rt][i]);
        }
    }
    __syncthreads();   // Xs dead from here; its region becomes x2s

    // ---- Layer 2: [32,512] x W_l1^T -> gelu -> x2s [32,256] --------------
    {
        v8f acc[2][2] = {};
        for (int k0 = 0; k0 < 512; k0 += 32) {
            v16h a0 = FRAG16(&x1s[(row)      * 512 + k0], kb);
            v16h a1 = FRAG16(&x1s[(row + 16) * 512 + k0], kb);
#pragma unroll
            for (int t = 0; t < 2; ++t) {
                int n = wave * 32 + t * 16 + nsel;
                v16h b = FRAG16(wpack + OFF_W1 + (size_t)n * 512 + k0, kb);
                acc[t][0] = __builtin_amdgcn_wmma_f32_16x16x32_f16(
                    false, a0, false, b, (short)0, acc[t][0], false, false);
                acc[t][1] = __builtin_amdgcn_wmma_f32_16x16x32_f16(
                    false, a1, false, b, (short)0, acc[t][1], false, false);
            }
        }
#pragma unroll
        for (int t = 0; t < 2; ++t) {
            int col = wave * 32 + t * 16 + nsel;
#pragma unroll
            for (int rt = 0; rt < 2; ++rt)
#pragma unroll
                for (int i = 0; i < 8; ++i)
                    x2s[(rt * 16 + i + rsh) * 256 + col] =
                        (_Float16)gelu_exact(acc[t][rt][i]);
        }
    }
    __syncthreads();   // x1s dead from here; its region becomes x3s/currs

    // ---- Layer 3: [32,256] x W_l2^T -> gelu -> x3s [32,128] --------------
    {
        v8f acc[2] = {};
        for (int k0 = 0; k0 < 256; k0 += 32) {
            v16h a0 = FRAG16(&x2s[(row)      * 256 + k0], kb);
            v16h a1 = FRAG16(&x2s[(row + 16) * 256 + k0], kb);
            int n = wave * 16 + nsel;
            v16h b = FRAG16(wpack + OFF_W2 + (size_t)n * 256 + k0, kb);
            acc[0] = __builtin_amdgcn_wmma_f32_16x16x32_f16(
                false, a0, false, b, (short)0, acc[0], false, false);
            acc[1] = __builtin_amdgcn_wmma_f32_16x16x32_f16(
                false, a1, false, b, (short)0, acc[1], false, false);
        }
        int col = wave * 16 + nsel;
#pragma unroll
        for (int rt = 0; rt < 2; ++rt)
#pragma unroll
            for (int i = 0; i < 8; ++i)
                x3s[(rt * 16 + i + rsh) * 128 + col] =
                    (_Float16)gelu_exact(acc[rt][i]);
    }
    __syncthreads();

    // ---- Layer 4: dot(128) -> sigmoid -> STE threshold -------------------
    if (tid < MTILE) {
        float s = 0.0f;
        for (int k = 0; k < 128; ++k)
            s += (float)x3s[tid * 128 + k] * w3f[k];
        float prob = 1.0f / (1.0f + expf(-s));
        float v    = prob * prev_m[rbase + tid];
        float curr = ((v > 0.5f) ? 1.0f : 0.0f) + 1e-10f;
        currs[tid]          = curr;
        currp[rbase + tid]  = curr;
        maskp[rbase + tid]  = (int)curr;   // 1 for 1+eps, 0 for eps
    }
    __syncthreads();

    // ---- out = input_feature * curr_m (full f32 precision; X is L2-hot) --
    {
        const v4f* src = (const v4f*)(X    + (size_t)rbase * 512);
        v4f*       dst = (v4f*)      (outp + (size_t)rbase * 512);
        for (int i = tid; i < (MTILE * 512) / 4; i += 256) {
            int   r = i >> 7;          // 128 float4 per row
            float c = currs[r];
            v4f v = src[i];
            v.x *= c; v.y *= c; v.z *= c; v.w *= c;
            dst[i] = v;
        }
    }
}

// ---------------------------------------------------------------------------
extern "C" void kernel_launch(void* const* d_in, const int* in_sizes, int n_in,
                              void* d_out, int out_size, void* d_ws, size_t ws_size,
                              hipStream_t stream) {
    (void)in_sizes; (void)n_in; (void)out_size; (void)ws_size;
    const float* X      = (const float*)d_in[0];  // input_feature [B,N,D]
    // d_in[1] = attention_mask (unused by the reference computation)
    const float* prev_m = (const float*)d_in[2];  // [B,N]
    const float* wL     = (const float*)d_in[3];  // [512,512]
    const float* w1     = (const float*)d_in[4];  // [256,512]
    const float* w2     = (const float*)d_in[5];  // [128,256]
    const float* w3     = (const float*)d_in[6];  // [1,128]

    _Float16* wpack = (_Float16*)d_ws;            // ~852 KB f16 weight pack

    float* outp  = (float*)d_out;                 // out   [B,N,D] f32
    int*   maskp = (int*)(outp + (size_t)ROWS * D_);   // mask  [B,N] i32
    float* currp = (float*)(maskp + ROWS);             // curr_m[B,N] f32

    convert_weights_f16<<<(TOT_W + 255) / 256, 256, 0, stream>>>(
        wL, w1, w2, w3, wpack);

    fused_mask_mlp<<<ROWS / MTILE, 256, 0, stream>>>(
        X, prev_m, w3, wpack, outp, maskp, currp);
}
